// Decoder2_36412732735703
// MI455X (gfx1250) — matrix operand order repaired and checked by reading, not compiled, not run on hardware
//
#include <hip/hip_runtime.h>

// ---------------------------------------------------------------------------
// Decoder block for MI455X (gfx1250): bf16 WMMA GEMMs + flash attention.
//   word_emb = causal_attn(wemb)      [2048,1024], H=16, d=64
//   cross    = attn(word_emb, pemb)   [2048,1024]
//   out      = relu(cross@W1^T+b1)@W2^T+b2   -> f32 [2048,1024]
// GEMM: 64x64 wave tile (16 WMMA : 16 b128 loads per K-step), templated
// epilogue (compile-time ldc -> immediate-offset stores).
// Attention: K/V^T tiles staged in LDS via async global->LDS copies
// (ASYNCcnt + s_wait_asynccnt) shared by the 8 waves of a block.
// ---------------------------------------------------------------------------

typedef __attribute__((ext_vector_type(16))) __bf16 bf16x16;
typedef __attribute__((ext_vector_type(8)))  __bf16 bf16x8;
typedef __attribute__((ext_vector_type(8)))  float  f32x8;
typedef int v4i_gl __attribute__((__vector_size__(16)));

#define DEV __device__ __forceinline__

// Async global->LDS staging (CDNA5 GLOBAL_LOAD_ASYNC_TO_LDS_B128) when the
// toolchain exposes it; synchronous copy fallback otherwise.
// Probe-derived signature: (global v4i32*, local v4i32*, imm offset, imm cpol)
#if defined(__AMDGCN__) && \
    __has_builtin(__builtin_amdgcn_global_load_async_to_lds_b128) && \
    __has_builtin(__builtin_amdgcn_s_wait_asynccnt)
#define ASYNC_LDS_COPY16(dst, src)                                          \
    __builtin_amdgcn_global_load_async_to_lds_b128(                         \
        (__attribute__((address_space(1))) v4i_gl*)(const void*)(src),      \
        (__attribute__((address_space(3))) v4i_gl*)(void*)(dst), 0, 0)
#define ASYNC_WAIT() __builtin_amdgcn_s_wait_asynccnt(0)
#else
#define ASYNC_LDS_COPY16(dst, src) (*(bf16x8*)(dst) = *(const bf16x8*)(src))
#define ASYNC_WAIT() ((void)0)
#endif

DEV f32x8 zero8() {
    f32x8 z;
#pragma unroll
    for (int i = 0; i < 8; ++i) z[i] = 0.f;
    return z;
}

DEV f32x8 wmma_bf16(bf16x16 a, bf16x16 b, f32x8 c) {
    // D = A(16x32) * B(32x16) + C, f32 accumulate
    return __builtin_amdgcn_wmma_f32_16x16x32_bf16(
        /*neg_a=*/false, a, /*neg_b=*/false, b,
        /*c_mod=*/(short)0, c, /*reuse_a=*/false, /*reuse_b=*/false);
}

// A-operand fragment, A row-major [M,K].
// ISA layout (16-bit A 16x32): lane<16: M=lane, K = k0+[0..7] & k0+[16..23]
//                              lane>=16: M=lane-16, K = k0+[8..15] & k0+[24..31]
DEV bf16x16 load_frag_a(const __bf16* A, int lda, int m0, int k0, int lane) {
    const int m  = m0 + (lane & 15);
    const int ks = (lane >> 4) << 3;           // 0 or 8
    const __bf16* p0 = A + (size_t)m * lda + k0 + ks;
    bf16x8 lo = *(const bf16x8*)p0;
    bf16x8 hi = *(const bf16x8*)(p0 + 16);
    bf16x16 r;
#pragma unroll
    for (int i = 0; i < 8; ++i) { r[i] = lo[i]; r[8 + i] = hi[i]; }
    return r;
}

// B-operand fragment for B stored row-major [N,K] (weight W[out,in] used as
// B^T).  ISA layout (16-bit B 32x16): lane<16: N=lane, K = k0+[0..7] & k0+[16..23];
// lane>=16: N=lane-16, K = k0+[8..15] & k0+[24..31].
DEV bf16x16 load_frag_b(const __bf16* B, int ldb, int n0, int k0, int lane) {
    const int n  = n0 + (lane & 15);
    const int kk = k0 + ((lane >> 4) << 3);    // 0 or 8
    const __bf16* p = B + (size_t)n * ldb + kk;
    bf16x8 lo = *(const bf16x8*)p;
    bf16x8 hi = *(const bf16x8*)(p + 16);
    bf16x16 r;
#pragma unroll
    for (int i = 0; i < 8; ++i) { r[i] = lo[i]; r[8 + i] = hi[i]; }
    return r;
}

DEV float lane_xor_f(float v, int mask, int lane) {
    const int addr = ((lane ^ mask) << 2);
    return __int_as_float(__builtin_amdgcn_ds_bpermute(addr, __float_as_int(v)));
}

// ---------------------------------------------------------------------------
// f32 -> bf16 conversion
// ---------------------------------------------------------------------------
__global__ __launch_bounds__(256) void k_cvt(const float* __restrict__ s,
                                             __bf16* __restrict__ d, int n) {
    int i = (blockIdx.x * 256 + threadIdx.x) * 4;
    if (i < n) {
        float4 f = *(const float4*)(s + i);
        d[i + 0] = (__bf16)f.x;
        d[i + 1] = (__bf16)f.y;
        d[i + 2] = (__bf16)f.z;
        d[i + 3] = (__bf16)f.w;
    }
}

// ---------------------------------------------------------------------------
// Generic NT GEMM: C[M,N] = A[M,K] @ B[N,K]^T + bias, optional ReLU.
// MODE: 0 = f32 out (ldc=LDC=N), 1 = bf16 out (ldc=LDC=N),
//       2 = bf16 transposed out C[n*LDC+m] (LDC=M) -> contiguous b128 stores.
// WG = 8 waves (4x2), WG tile 256x128, wave tile 64x64.
// M % 256 == 0, N % 128 == 0, K % 32 == 0 (true for all calls here).
// ---------------------------------------------------------------------------
template <int LDC, int MODE, int RELU>
__global__ __launch_bounds__(256) void k_gemm_nt(
    const __bf16* __restrict__ A, const __bf16* __restrict__ B,
    const float* __restrict__ bias, void* __restrict__ Cv,
    int M, int N, int K)
{
    const int lane = threadIdx.x & 31;
    const int wid  = __builtin_amdgcn_readfirstlane(threadIdx.x >> 5);
    const int m0 = blockIdx.y * 256 + (wid >> 1) * 64;
    const int n0 = blockIdx.x * 128 + (wid & 1) * 64;

    f32x8 acc[4][4];
#pragma unroll
    for (int mi = 0; mi < 4; ++mi)
#pragma unroll
        for (int ni = 0; ni < 4; ++ni) acc[mi][ni] = zero8();

    for (int k0 = 0; k0 < K; k0 += 32) {
        bf16x16 a[4], b[4];
#pragma unroll
        for (int mi = 0; mi < 4; ++mi)
            a[mi] = load_frag_a(A, K, m0 + 16 * mi, k0, lane);
#pragma unroll
        for (int ni = 0; ni < 4; ++ni)
            b[ni] = load_frag_b(B, K, n0 + 16 * ni, k0, lane);
        if (k0 + 32 < K) {   // pull next K-tile toward caches (global_prefetch_b8)
            __builtin_prefetch(A + (size_t)(m0 + (lane & 15)) * K + k0 + 32, 0, 3);
            __builtin_prefetch(B + (size_t)(n0 + (lane & 15)) * K + k0 + 32, 0, 3);
        }
#pragma unroll
        for (int mi = 0; mi < 4; ++mi)
#pragma unroll
            for (int ni = 0; ni < 4; ++ni)
                acc[mi][ni] = wmma_bf16(a[mi], b[ni], acc[mi][ni]);
    }

    const int nlo = lane & 15, half = lane >> 4;
#pragma unroll
    for (int mi = 0; mi < 4; ++mi)
#pragma unroll
        for (int ni = 0; ni < 4; ++ni) {
            const int col = n0 + ni * 16 + nlo;
            const float bv = bias[col];
            const int rb = m0 + mi * 16 + half * 8;
            if (MODE == 2) {
                // transposed: 8 consecutive rows are contiguous -> one b128
                __bf16* p = (__bf16*)Cv + (size_t)col * LDC + rb;
                bf16x8 vs;
#pragma unroll
                for (int i = 0; i < 8; ++i) {
                    float v = acc[mi][ni][i] + bv;
                    if (RELU) v = fmaxf(v, 0.f);
                    vs[i] = (__bf16)v;
                }
                *(bf16x8*)p = vs;
            } else if (MODE == 1) {
                __bf16* p = (__bf16*)Cv + (size_t)rb * LDC + col;
#pragma unroll
                for (int i = 0; i < 8; ++i) {
                    float v = acc[mi][ni][i] + bv;
                    if (RELU) v = fmaxf(v, 0.f);
                    p[i * LDC] = (__bf16)v;   // compile-time immediate offsets
                }
            } else {
                float* p = (float*)Cv + (size_t)rb * LDC + col;
#pragma unroll
                for (int i = 0; i < 8; ++i) {
                    float v = acc[mi][ni][i] + bv;
                    if (RELU) v = fmaxf(v, 0.f);
                    p[i * LDC] = v;
                }
            }
        }
}

// ---------------------------------------------------------------------------
// Flash attention (H=16, d=64, feature ld = 1024).  Block = 8 waves; each wave
// owns a 16-row query block of one head.  The 8 waves share K/V^T tiles staged
// cooperatively in LDS via async global->LDS copies; online softmax per wave.
// Q,K: bf16 [S,1024]; VT: bf16 [1024, Sk] (V transposed); Out: bf16 [Sq,1024].
// grid = (Sq/128, H), block = 256.
// ---------------------------------------------------------------------------
__global__ __launch_bounds__(256) void k_flash_attn(
    const __bf16* __restrict__ Q, const __bf16* __restrict__ Kt,
    const __bf16* __restrict__ VT, __bf16* __restrict__ Out,
    int Sq, int Sk, int causal)
{
    __shared__ alignas(16) __bf16 sK[32 * 64];      // keys x feat
    __shared__ alignas(16) __bf16 sV[64 * 32];      // feat x keys (V^T tile)
    __shared__ alignas(16) __bf16 sP[8][16 * 32];   // per-wave P staging

    const int tid  = threadIdx.x;
    const int lane = tid & 31;
    const int wid  = __builtin_amdgcn_readfirstlane(tid >> 5);
    const int bm0  = blockIdx.x * 128;
    const int m0   = bm0 + wid * 16;
    const int hoff = blockIdx.y * 64;
    const int ld   = 1024;
    const float scale = 0.125f;                     // 1/sqrt(64)

    // Q fragments held for the whole pass (d=64 -> two K=32 steps)
    bf16x16 qa0 = load_frag_a(Q + hoff, ld, m0, 0,  lane);
    bf16x16 qa1 = load_frag_a(Q + hoff, ld, m0, 32, lane);

    f32x8 o0 = zero8(), o1 = zero8(), o2 = zero8(), o3 = zero8();
    f32x8 mrow, lrow;
#pragma unroll
    for (int i = 0; i < 8; ++i) { mrow[i] = -1e30f; lrow[i] = 0.f; }

    const int nlo = lane & 15, half = lane >> 4;
    const int rbase = m0 + half * 8;
    __bf16* pb = &sP[wid][0];
    const int tend  = causal ? (m0 + 16) : Sk;      // wave-uniform (scalar)
    const int btend = causal ? (bm0 + 128) : Sk;    // block-uniform

    // cooperative staging indices (one 16B async copy per thread per tile)
    const int kr = tid >> 3, kc = (tid & 7) * 8;    // K tile [32][64]
    const int vr = tid >> 2, vc = (tid & 3) * 8;    // V^T tile [64][32]

    for (int t0 = 0; t0 < btend; t0 += 32) {
        // stage K (rows t0..t0+31, feats hoff..hoff+63) and V^T (feats, keys)
        ASYNC_LDS_COPY16(sK + kr * 64 + kc,
                         Kt + (size_t)(t0 + kr) * ld + hoff + kc);
        ASYNC_LDS_COPY16(sV + vr * 32 + vc,
                         VT + (size_t)(hoff + vr) * Sk + t0 + vc);
        ASYNC_WAIT();
        __syncthreads();

        if (t0 < tend) {                            // scalar branch per wave
            // S = Q K^T for 32 keys: two 16-col fragments
            f32x8 s0 = zero8(), s1 = zero8();
            {
                bf16x16 kb;
                kb = load_frag_b(sK, 64, 0,  0,  lane); s0 = wmma_bf16(qa0, kb, s0);
                kb = load_frag_b(sK, 64, 0,  32, lane); s0 = wmma_bf16(qa1, kb, s0);
                kb = load_frag_b(sK, 64, 16, 0,  lane); s1 = wmma_bf16(qa0, kb, s1);
                kb = load_frag_b(sK, 64, 16, 32, lane); s1 = wmma_bf16(qa1, kb, s1);
            }
            const int c0 = t0 + nlo, c1 = t0 + 16 + nlo;
            f32x8 tmax;
#pragma unroll
            for (int i = 0; i < 8; ++i) {
                float v0 = s0[i] * scale;
                float v1 = s1[i] * scale;
                if (causal) {
                    const int r = rbase + i;
                    if (c0 > r) v0 = -1e30f;
                    if (c1 > r) v1 = -1e30f;
                }
                s0[i] = v0; s1[i] = v1;
                tmax[i] = fmaxf(v0, v1);
            }
            // row-max across the 16 lanes of each half
#pragma unroll
            for (int d = 1; d < 16; d <<= 1)
#pragma unroll
                for (int i = 0; i < 8; ++i)
                    tmax[i] = fmaxf(tmax[i], lane_xor_f(tmax[i], d, lane));

            f32x8 alpha, rs;
#pragma unroll
            for (int i = 0; i < 8; ++i) {
                const float mn = fmaxf(mrow[i], tmax[i]);
                alpha[i] = __expf(mrow[i] - mn);
                mrow[i]  = mn;
                const float p0 = __expf(s0[i] - mn);
                const float p1 = __expf(s1[i] - mn);
                s0[i] = p0; s1[i] = p1;
                rs[i] = p0 + p1;
            }
#pragma unroll
            for (int d = 1; d < 16; d <<= 1)
#pragma unroll
                for (int i = 0; i < 8; ++i)
                    rs[i] += lane_xor_f(rs[i], d, lane);

#pragma unroll
            for (int i = 0; i < 8; ++i) {
                lrow[i] = lrow[i] * alpha[i] + rs[i];
                o0[i] *= alpha[i]; o1[i] *= alpha[i];
                o2[i] *= alpha[i]; o3[i] *= alpha[i];
            }

            // Stage P (16x32, bf16) in wave-private LDS: C-layout -> A-layout
#pragma unroll
            for (int i = 0; i < 8; ++i) {
                pb[(half * 8 + i) * 32 + nlo]      = (__bf16)s0[i];
                pb[(half * 8 + i) * 32 + 16 + nlo] = (__bf16)s1[i];
            }
            __builtin_amdgcn_wave_barrier();
            bf16x16 pa = load_frag_a(pb, 32, 0, 0, lane);

            // O += P @ V : V^T tile rows are features, contiguous along keys
            bf16x16 vb0 = load_frag_b(sV, 32,  0, 0, lane);
            bf16x16 vb1 = load_frag_b(sV, 32, 16, 0, lane);
            bf16x16 vb2 = load_frag_b(sV, 32, 32, 0, lane);
            bf16x16 vb3 = load_frag_b(sV, 32, 48, 0, lane);
            o0 = wmma_bf16(pa, vb0, o0);
            o1 = wmma_bf16(pa, vb1, o1);
            o2 = wmma_bf16(pa, vb2, o2);
            o3 = wmma_bf16(pa, vb3, o3);
        }
        __syncthreads();
    }

#pragma unroll
    for (int i = 0; i < 8; ++i) {
        const float inv = 1.f / lrow[i];
        const size_t r = (size_t)(rbase + i) * ld + hoff + nlo;
        Out[r +  0] = (__bf16)(o0[i] * inv);
        Out[r + 16] = (__bf16)(o1[i] * inv);
        Out[r + 32] = (__bf16)(o2[i] * inv);
        Out[r + 48] = (__bf16)(o3[i] * inv);
    }
}

// ---------------------------------------------------------------------------
extern "C" void kernel_launch(void* const* d_in, const int* in_sizes, int n_in,
                              void* d_out, int out_size, void* d_ws, size_t ws_size,
                              hipStream_t stream) {
    (void)in_sizes; (void)n_in; (void)out_size; (void)ws_size;
    const float* wemb = (const float*)d_in[0];
    const float* pemb = (const float*)d_in[1];
    const float* Wq_m = (const float*)d_in[2];  const float* bq_m = (const float*)d_in[3];
    const float* Wk_m = (const float*)d_in[4];  const float* bk_m = (const float*)d_in[5];
    const float* Wv_m = (const float*)d_in[6];  const float* bv_m = (const float*)d_in[7];
    const float* Wq_c = (const float*)d_in[8];  const float* bq_c = (const float*)d_in[9];
    const float* Wk_c = (const float*)d_in[10]; const float* bk_c = (const float*)d_in[11];
    const float* Wv_c = (const float*)d_in[12]; const float* bv_c = (const float*)d_in[13];
    const float* W1   = (const float*)d_in[14]; const float* b1   = (const float*)d_in[15];
    const float* W2   = (const float*)d_in[16]; const float* b2   = (const float*)d_in[17];
    float* out = (float*)d_out;

    const int SW = 2048, SP = 1024, DM = 1024, DFF = 4096;

    __bf16* ws = (__bf16*)d_ws;
    size_t off = 0;
    auto take = [&](size_t n) { __bf16* p = ws + off; off += n; return p; };
    __bf16* Xw   = take((size_t)SW * DM);   // bf16 wemb
    __bf16* Xp   = take((size_t)SP * DM);   // bf16 pemb
    __bf16* hWqm = take((size_t)DM * DM);
    __bf16* hWkm = take((size_t)DM * DM);
    __bf16* hWvm = take((size_t)DM * DM);
    __bf16* hWqc = take((size_t)DM * DM);
    __bf16* hWkc = take((size_t)DM * DM);
    __bf16* hWvc = take((size_t)DM * DM);
    __bf16* hW1  = take((size_t)DFF * DM);
    __bf16* hW2  = take((size_t)DM * DFF);
    __bf16* Qm   = take((size_t)SW * DM);
    __bf16* Km   = take((size_t)SW * DM);
    __bf16* VmT  = take((size_t)SW * DM);   // V_self transposed [DM, SW]
    __bf16* Wrd  = take((size_t)SW * DM);   // word_emb
    __bf16* Qc   = take((size_t)SW * DM);
    __bf16* Kc   = take((size_t)SP * DM);
    __bf16* VcT  = take((size_t)SP * DM);   // V_cross transposed [DM, SP]
    __bf16* Xc   = take((size_t)SW * DM);   // cross_emb
    __bf16* Hf   = take((size_t)SW * DFF);  // FFN hidden

    auto cvt = [&](const float* s, __bf16* d, int n) {
        k_cvt<<<dim3(n / 1024), dim3(256), 0, stream>>>(s, d, n);
    };
    cvt(wemb, Xw, SW * DM);
    cvt(pemb, Xp, SP * DM);
    cvt(Wq_m, hWqm, DM * DM);
    cvt(Wk_m, hWkm, DM * DM);
    cvt(Wv_m, hWvm, DM * DM);
    cvt(Wq_c, hWqc, DM * DM);
    cvt(Wk_c, hWkc, DM * DM);
    cvt(Wv_c, hWvc, DM * DM);
    cvt(W1,   hW1,  DFF * DM);
    cvt(W2,   hW2,  DM * DFF);

    // Self-attention projections (V stored transposed for the PV matmul)
    k_gemm_nt<1024, 1, 0><<<dim3(8, 8),  256, 0, stream>>>(Xw, hWqm, bq_m, Qm,  SW, DM, DM);
    k_gemm_nt<1024, 1, 0><<<dim3(8, 8),  256, 0, stream>>>(Xw, hWkm, bk_m, Km,  SW, DM, DM);
    k_gemm_nt<2048, 2, 0><<<dim3(8, 8),  256, 0, stream>>>(Xw, hWvm, bv_m, VmT, SW, DM, DM);
    // Causal flash attention -> word_emb
    k_flash_attn<<<dim3(SW / 128, 16), dim3(256), 0, stream>>>(
        Qm, Km, VmT, Wrd, SW, SW, 1);
    // Cross-attention projections
    k_gemm_nt<1024, 1, 0><<<dim3(8, 8),  256, 0, stream>>>(Wrd, hWqc, bq_c, Qc,  SW, DM, DM);
    k_gemm_nt<1024, 1, 0><<<dim3(8, 4),  256, 0, stream>>>(Xp,  hWkc, bk_c, Kc,  SP, DM, DM);
    k_gemm_nt<1024, 2, 0><<<dim3(8, 4),  256, 0, stream>>>(Xp,  hWvc, bv_c, VcT, SP, DM, DM);
    // Cross flash attention -> cross_emb
    k_flash_attn<<<dim3(SW / 128, 16), dim3(256), 0, stream>>>(
        Qc, Kc, VcT, Xc, SW, SP, 0);
    // FFN
    k_gemm_nt<4096, 1, 1><<<dim3(32, 8), 256, 0, stream>>>(Xc, hW1, b1, Hf, SW, DFF, DM);
    k_gemm_nt<1024, 0, 0><<<dim3(8, 8),  256, 0, stream>>>(Hf, hW2, b2, out, SW, DM, DFF);
}
